// GRAFFNet_42056319762544
// MI455X (gfx1250) — matrix operand order repaired and checked by reading, not compiled, not run on hardware
//
#include <hip/hip_runtime.h>
#include <math.h>

typedef __attribute__((ext_vector_type(16))) _Float16 v16h;
typedef __attribute__((ext_vector_type(8)))  float    v8f;

#define NNODES 50000
#define NEDGES 800000
#define NFEAT  500
#define NHID   64
#define NCLASS 40
#define STEPSZ 0.1f

#define KTILES_ENC 16   // 512 padded K (15 full + 1 tail)
#define KTILES_SYM 2    // K = 64

// ---------------- B-fragment pre-packing --------------------------------------
// Packed layout: pb[(((t*2 + hi)*64 + n)*16 + i]  (f16)
//   = B[k = t*32 + (i < 8 ? hi*8 + i : 16 + hi*8 + (i-8))][n], zero-padded past K.
// In the GEMM, lane (hi,n) loads its 16-half fragment as one contiguous 32B vec.

__global__ void k_pack_enc(const float* __restrict__ Wenc, _Float16* __restrict__ pb) {
    int idx = blockIdx.x * blockDim.x + threadIdx.x;       // 16*2*64*16 = 32768
    if (idx >= KTILES_ENC * 2 * 64 * 16) return;
    int i  = idx & 15;
    int n  = (idx >> 4) & 63;
    int hi = (idx >> 10) & 1;
    int t  = idx >> 11;
    int k  = t * 32 + (i < 8 ? hi * 8 + i : 16 + hi * 8 + (i - 8));
    pb[idx] = (k < NFEAT) ? (_Float16)Wenc[k * NHID + n] : (_Float16)0.0f;
}

__global__ void k_pack_sym(const float* __restrict__ Wpair, _Float16* __restrict__ pb) {
    int idx = blockIdx.x * blockDim.x + threadIdx.x;       // 2*2*64*16 = 4096
    if (idx >= KTILES_SYM * 2 * 64 * 16) return;
    int i  = idx & 15;
    int n  = (idx >> 4) & 63;
    int hi = (idx >> 10) & 1;
    int t  = idx >> 11;
    int k  = t * 32 + (i < 8 ? hi * 8 + i : 16 + hi * 8 + (i - 8)); // always < 64
    pb[idx] = (_Float16)(0.5f * (Wpair[k * NHID + n] + Wpair[n * NHID + k]));
}

// ---------------- degree / normalization --------------------------------------

__global__ void k_deg_init(float* __restrict__ deg) {
    int i = blockIdx.x * blockDim.x + threadIdx.x;
    if (i < NNODES) deg[i] = 1.0f;   // self loop contributes 1
}

__global__ void k_deg_accum(const int* __restrict__ ei, float* __restrict__ deg) {
    int e = blockIdx.x * blockDim.x + threadIdx.x;
    if (e < NEDGES) atomicAdd(&deg[ei[e]], 1.0f);           // row = dst = edge_index[0]
}

__global__ void k_dinv(const float* __restrict__ deg, float* __restrict__ dinv) {
    int i = blockIdx.x * blockDim.x + threadIdx.x;
    if (i < NNODES) dinv[i] = rsqrtf(fmaxf(deg[i], 1.0f));
}

// ---------------- fragment helpers --------------------------------------------

__device__ __forceinline__ v16h cvt_a_frag(float4 a0, float4 a1, float4 a2, float4 a3) {
    v16h a;
    a[0]  = (_Float16)a0.x; a[1]  = (_Float16)a0.y; a[2]  = (_Float16)a0.z; a[3]  = (_Float16)a0.w;
    a[4]  = (_Float16)a1.x; a[5]  = (_Float16)a1.y; a[6]  = (_Float16)a1.z; a[7]  = (_Float16)a1.w;
    a[8]  = (_Float16)a2.x; a[9]  = (_Float16)a2.y; a[10] = (_Float16)a2.z; a[11] = (_Float16)a2.w;
    a[12] = (_Float16)a3.x; a[13] = (_Float16)a3.y; a[14] = (_Float16)a3.z; a[15] = (_Float16)a3.w;
    return a;
}

// ---------------- encoder GEMM: h = x @ W_enc  (f16 WMMA, f32 accum) -----------
// block = 128 = 4 waves; block owns 16 rows, wave w owns cols [16w,16w+16)

__global__ __launch_bounds__(128)
void k_encoder(const float* __restrict__ x, const _Float16* __restrict__ pb,
               float* __restrict__ h, float* __restrict__ h0) {
    const int lane = threadIdx.x & 31;
    const int wave = threadIdx.x >> 5;
    const int m    = lane & 15;
    const int hi   = lane >> 4;
    const int row0 = blockIdx.x * 16;
    const int n0   = wave * 16;
    const int n    = n0 + m;

    const float*    xrow = &x[(size_t)(row0 + m) * NFEAT];
    const _Float16* pbn  = &pb[(size_t)(hi * 64 + n) * 16];

    v8f c = {};
    #pragma unroll 1
    for (int t = 0; t < 15; ++t) {                     // guard-free full K tiles
        const int k0 = t * 32;
        __builtin_prefetch(&xrow[k0 + 64], 0, 1);      // gfx1250 global_prefetch_b8
        float4 a0 = *(const float4*)&xrow[k0 + hi * 8];
        float4 a1 = *(const float4*)&xrow[k0 + hi * 8 + 4];
        float4 a2 = *(const float4*)&xrow[k0 + 16 + hi * 8];
        float4 a3 = *(const float4*)&xrow[k0 + 16 + hi * 8 + 4];
        v16h a = cvt_a_frag(a0, a1, a2, a3);
        v16h b = *(const v16h*)&pbn[(size_t)t * 2 * 64 * 16];
        c = __builtin_amdgcn_wmma_f32_16x16x32_f16(false, a, false, b,
                                                   (short)0, c, false, false);
    }
    {   // tail tile: K 480..499 valid, 500..511 zero
        float4 a0 = *(const float4*)&xrow[480 + hi * 8];
        float4 a1 = *(const float4*)&xrow[484 + hi * 8];
        float4 a2 = make_float4(0.f, 0.f, 0.f, 0.f);
        if (hi == 0) a2 = *(const float4*)&xrow[496];  // K 496..499
        float4 a3 = make_float4(0.f, 0.f, 0.f, 0.f);
        v16h a = cvt_a_frag(a0, a1, a2, a3);
        v16h b = *(const v16h*)&pbn[(size_t)15 * 2 * 64 * 16];
        c = __builtin_amdgcn_wmma_f32_16x16x32_f16(false, a, false, b,
                                                   (short)0, c, false, false);
    }
    #pragma unroll
    for (int r = 0; r < 8; ++r) {                      // C: VGPR r -> row r / r+8
        int rr = row0 + r + hi * 8;
        float v = c[r];
        h [(size_t)rr * NHID + n] = v;
        h0[(size_t)rr * NHID + n] = v;
    }
}

// ---------------- pairwise GEMM: m = h @ W_sym ---------------------------------

__global__ __launch_bounds__(128)
void k_pair(const float* __restrict__ hcur, const _Float16* __restrict__ pb,
            float* __restrict__ mout) {
    const int lane = threadIdx.x & 31;
    const int wave = threadIdx.x >> 5;
    const int m    = lane & 15;
    const int hi   = lane >> 4;
    const int row0 = blockIdx.x * 16;
    const int n0   = wave * 16;
    const int n    = n0 + m;

    const float*    hrow = &hcur[(size_t)(row0 + m) * NHID];
    const _Float16* pbn  = &pb[(size_t)(hi * 64 + n) * 16];

    v8f c = {};
    #pragma unroll
    for (int t = 0; t < KTILES_SYM; ++t) {
        const int k0 = t * 32;
        float4 a0 = *(const float4*)&hrow[k0 + hi * 8];
        float4 a1 = *(const float4*)&hrow[k0 + hi * 8 + 4];
        float4 a2 = *(const float4*)&hrow[k0 + 16 + hi * 8];
        float4 a3 = *(const float4*)&hrow[k0 + 16 + hi * 8 + 4];
        v16h a = cvt_a_frag(a0, a1, a2, a3);
        v16h b = *(const v16h*)&pbn[(size_t)t * 2 * 64 * 16];
        c = __builtin_amdgcn_wmma_f32_16x16x32_f16(false, a, false, b,
                                                   (short)0, c, false, false);
    }
    #pragma unroll
    for (int r = 0; r < 8; ++r) {
        int rr = row0 + r + hi * 8;
        mout[(size_t)rr * NHID + n] = c[r];
    }
}

// ---------------- normalized scatter-add (edges + self loops) ------------------
// 16 lanes per entry, float4 gather, f32 atomic scatter

__global__ void k_scatter(const int* __restrict__ ei, const float* __restrict__ dinv,
                          const float* __restrict__ msg, float* __restrict__ agg) {
    long long t   = (long long)blockIdx.x * blockDim.x + threadIdx.x;
    long long idx = t >> 4;
    int f = (int)(t & 15) * 4;
    if (idx >= (long long)(NEDGES + NNODES)) return;
    int r, cidx; float nrm;
    if (idx < NEDGES) {
        r    = ei[idx];                 // dst
        cidx = ei[NEDGES + idx];        // src
        nrm  = dinv[r] * dinv[cidx];
    } else {
        r = cidx = (int)(idx - NEDGES); // self loop
        float d = dinv[r];
        nrm = d * d;
    }
    const float4 v = *(const float4*)&msg[(size_t)cidx * NHID + f];
    float* dst = &agg[(size_t)r * NHID + f];
    atomicAdd(dst + 0, v.x * nrm);
    atomicAdd(dst + 1, v.y * nrm);
    atomicAdd(dst + 2, v.z * nrm);
    atomicAdd(dst + 3, v.w * nrm);
}

// ---------------- ODE step: h += STEP * (agg - h*omega - beta*h0) --------------

__global__ void k_update(float* __restrict__ h, const float* __restrict__ agg,
                         const float* __restrict__ omega, const float* __restrict__ beta,
                         const float* __restrict__ h0) {
    long long i = (long long)blockIdx.x * blockDim.x + threadIdx.x;
    if (i < (long long)NNODES * NHID) {
        int f = (int)(i & (NHID - 1));
        float hv   = h[i];
        float conv = agg[i] - hv * omega[f] - beta[0] * h0[i];
        h[i] = fmaf(STEPSZ, conv, hv);
    }
}

// ---------------- decoder + log_softmax ----------------------------------------

__global__ __launch_bounds__(256)
void k_decoder(const float* __restrict__ h, const float* __restrict__ Wdec,
               float* __restrict__ out) {
    __shared__ float w[NHID * NCLASS];
    for (int i = threadIdx.x; i < NHID * NCLASS; i += blockDim.x) w[i] = Wdec[i];
    __syncthreads();
    int node = blockIdx.x * blockDim.x + threadIdx.x;
    if (node >= NNODES) return;
    float acc[NCLASS];
    #pragma unroll
    for (int c = 0; c < NCLASS; ++c) acc[c] = 0.0f;
    const float* hr = &h[(size_t)node * NHID];
    for (int k = 0; k < NHID; ++k) {
        float hk = hr[k];
        #pragma unroll
        for (int c = 0; c < NCLASS; ++c)
            acc[c] = fmaf(hk, w[k * NCLASS + c], acc[c]);
    }
    float mx = acc[0];
    #pragma unroll
    for (int c = 1; c < NCLASS; ++c) mx = fmaxf(mx, acc[c]);
    float s = 0.0f;
    #pragma unroll
    for (int c = 0; c < NCLASS; ++c) s += __expf(acc[c] - mx);
    float lse = mx + __logf(s);
    float* orow = &out[(size_t)node * NCLASS];
    #pragma unroll
    for (int c = 0; c < NCLASS; ++c) orow[c] = acc[c] - lse;
}

// ---------------- launcher ------------------------------------------------------

extern "C" void kernel_launch(void* const* d_in, const int* in_sizes, int n_in,
                              void* d_out, int out_size, void* d_ws, size_t ws_size,
                              hipStream_t stream) {
    (void)in_sizes; (void)n_in; (void)out_size; (void)ws_size;

    const float* x     = (const float*)d_in[0];
    const float* Wenc  = (const float*)d_in[1];
    const float* omega = (const float*)d_in[2];
    const float* Wpair = (const float*)d_in[3];
    const float* beta  = (const float*)d_in[4];
    const float* Wdec  = (const float*)d_in[5];
    const int*   ei    = (const int*)d_in[6];
    float* out = (float*)d_out;

    char* ws = (char*)d_ws;
    float* deg  = (float*)ws; ws += sizeof(float) * NNODES;
    float* dinv = (float*)ws; ws += sizeof(float) * NNODES;
    float* h    = (float*)ws; ws += sizeof(float) * (size_t)NNODES * NHID;
    float* h0   = (float*)ws; ws += sizeof(float) * (size_t)NNODES * NHID;
    float* msg  = (float*)ws; ws += sizeof(float) * (size_t)NNODES * NHID;
    float* agg  = (float*)ws; ws += sizeof(float) * (size_t)NNODES * NHID;
    _Float16* pbE = (_Float16*)ws; ws += sizeof(_Float16) * KTILES_ENC * 2 * 64 * 16;
    _Float16* pbS = (_Float16*)ws; ws += sizeof(_Float16) * KTILES_SYM * 2 * 64 * 16;

    // one-time weight packing into WMMA B-fragment order
    k_pack_enc<<<(KTILES_ENC * 2 * 64 * 16 + 255) / 256, 256, 0, stream>>>(Wenc, pbE);
    k_pack_sym<<<(KTILES_SYM * 2 * 64 * 16 + 255) / 256, 256, 0, stream>>>(Wpair, pbS);

    // GCN normalization: deg = 1 + in-degree; dinv = rsqrt(max(deg,1))
    k_deg_init <<<(NNODES + 255) / 256, 256, 0, stream>>>(deg);
    k_deg_accum<<<(NEDGES + 255) / 256, 256, 0, stream>>>(ei, deg);
    k_dinv     <<<(NNODES + 255) / 256, 256, 0, stream>>>(deg, dinv);

    // encoder
    k_encoder<<<NNODES / 16, 128, 0, stream>>>(x, pbE, h, h0);

    // two GRAFF layers
    for (int layer = 0; layer < 2; ++layer) {
        k_pair<<<NNODES / 16, 128, 0, stream>>>(h, pbS, msg);
        hipMemsetAsync(agg, 0, sizeof(float) * (size_t)NNODES * NHID, stream);
        long long tot = (long long)(NEDGES + NNODES) * 16;
        k_scatter<<<(int)((tot + 255) / 256), 256, 0, stream>>>(ei, dinv, msg, agg);
        long long elems = (long long)NNODES * NHID;
        k_update<<<(int)((elems + 255) / 256), 256, 0, stream>>>(h, agg, omega, beta, h0);
    }

    // decoder + log_softmax
    k_decoder<<<(NNODES + 255) / 256, 256, 0, stream>>>(h, Wdec, out);
}
